// MyAttention_51805895524899
// MI455X (gfx1250) — matrix operand order repaired and checked by reading, not compiled, not run on hardware
//
#include <hip/hip_runtime.h>
#include <hip/hip_bf16.h>

// ---------------------------------------------------------------------------
// Problem constants (fixed by the reference): B=4, S=8192, D=1024, H=16,
// d_k=64, chunk P=128, C=64 chunks.
// ---------------------------------------------------------------------------
#define D_MODEL 1024
#define NROWS   32768          // B*S
#define D_K     64
#define P_CHUNK 128

typedef __attribute__((ext_vector_type(16))) __bf16 v16bf;
typedef __attribute__((ext_vector_type(8)))  __bf16 v8bf;
typedef __attribute__((ext_vector_type(8)))  float  v8f;
typedef __attribute__((ext_vector_type(4)))  float  v4f;

union BF16Frag { v16bf v; v8bf h[2]; };

__device__ __forceinline__ v8f wmma_bf16(v16bf a, v16bf b, v8f c) {
    // D = A(16x32 bf16) * B(32x16 bf16) + C(16x16 f32)
    return __builtin_amdgcn_wmma_f32_16x16x32_bf16(
        /*neg_a=*/false, a, /*neg_b=*/false, b,
        /*c_mod=*/(short)0, c, /*reuse_a=*/false, /*reuse_b=*/false);
}

__device__ __forceinline__ v8f zero8() {
    v8f z = {0.f, 0.f, 0.f, 0.f, 0.f, 0.f, 0.f, 0.f};
    return z;
}

// ---- CDNA5 async global->LDS copy (ASYNCcnt path, 16B per lane) -----------
typedef __attribute__((address_space(3))) char lds_char_t;

__device__ __forceinline__ unsigned lds_byte_off(const void* p) {
    return (unsigned)(unsigned long long)(lds_char_t*)p;
}

__device__ __forceinline__ void async_copy_b128(unsigned lds_off, const void* gptr) {
    // GLOBAL_LOAD_ASYNC_TO_LDS_B128: copies 16B from global memory into LDS at offset vdst, tracked by ASYNCcnt
    asm volatile("global_load_async_to_lds_b128 %0, %1, off"
                 :
                 : "v"(lds_off), "v"(gptr)
                 : "memory");
}

__device__ __forceinline__ void wait_asynccnt0() {
    asm volatile("s_wait_asynccnt 0x0" ::: "memory");
}

// ---------------------------------------------------------------------------
// Y[M,1024] = A[M,1024] @ W^T + bias    (W is [1024,1024] row-major, fp32)
// A is fp32 (A_BF16=0) or bf16 (A_BF16=1); Y is bf16 (OUT_BF16=1) or fp32.
// Block tile 128(m) x 256(n); 8 waves as 2(m) x 4(n), each wave a 64x64 tile
// (4x4 WMMA accumulators). K-loop in steps of 32 (one 16x16x32 WMMA deep).
// ---------------------------------------------------------------------------
template <bool A_BF16, bool OUT_BF16>
__global__ __launch_bounds__(256)
void gemm_bias_kernel(const void* __restrict__ Ap,
                      const float* __restrict__ W,
                      const float* __restrict__ bias,
                      void* __restrict__ Yp) {
    constexpr int K = D_MODEL;
    constexpr int N = D_MODEL;

    __shared__ __attribute__((aligned(16))) __bf16 As[128][48]; // 32 + pad, 96B rows
    __shared__ __attribute__((aligned(16))) __bf16 Bs[256][48];

    const int tid  = threadIdx.x;
    const int lane = tid & 31;
    const int wave = tid >> 5;
    const int r    = lane & 15;   // row/col-in-tile selector
    const int lh   = lane >> 4;   // lane half
    const int wm   = wave >> 2;   // 0..1
    const int wn   = wave & 3;    // 0..3

    const long mbase = (long)blockIdx.y * 128;
    const int  nbase = blockIdx.x * 256;

    v8f acc[4][4];
    for (int i = 0; i < 4; ++i)
        for (int j = 0; j < 4; ++j) acc[i][j] = zero8();

    for (int kk = 0; kk < K; kk += 32) {
        // ---- stage A tile (128 x 32) into LDS as bf16 ----
        if (A_BF16) {
            const __bf16* A = (const __bf16*)Ap;
            for (int pass = 0; pass < 2; ++pass) {
                int row = pass * 64 + (tid >> 2);
                int ck  = (tid & 3) * 8;
                async_copy_b128(lds_byte_off(&As[row][ck]),
                                &A[(mbase + row) * K + kk + ck]);
            }
        } else {
            const float* A = (const float*)Ap;
            for (int pass = 0; pass < 4; ++pass) {
                int row = pass * 32 + (tid >> 3);
                int ck  = (tid & 7) * 4;
                const float* src = &A[(mbase + row) * K + kk + ck];
                if (kk + 32 < K) __builtin_prefetch(src + 32, 0, 1);
                v4f x = *(const v4f*)src;
                As[row][ck + 0] = (__bf16)x.x;
                As[row][ck + 1] = (__bf16)x.y;
                As[row][ck + 2] = (__bf16)x.z;
                As[row][ck + 3] = (__bf16)x.w;
            }
        }
        // ---- stage W tile (rows nbase..nbase+255, cols kk..kk+31) ----
        for (int pass = 0; pass < 8; ++pass) {
            int row = pass * 32 + (tid >> 3);
            int ck  = (tid & 7) * 4;
            const float* src = &W[(long)(nbase + row) * K + kk + ck];
            if (kk + 32 < K) __builtin_prefetch(src + 32, 0, 1);
            v4f x = *(const v4f*)src;
            Bs[row][ck + 0] = (__bf16)x.x;
            Bs[row][ck + 1] = (__bf16)x.y;
            Bs[row][ck + 2] = (__bf16)x.z;
            Bs[row][ck + 3] = (__bf16)x.w;
        }
        if (A_BF16) wait_asynccnt0();
        __syncthreads();

        // ---- B fragments (lane = col n, contiguous 32B at K=16*lh) ----
        BF16Frag bfrag[4];
        for (int ni = 0; ni < 4; ++ni) {
            int brow = wn * 64 + ni * 16 + r;
            bfrag[ni].h[0] = *(const v8bf*)&Bs[brow][16 * lh];
            bfrag[ni].h[1] = *(const v8bf*)&Bs[brow][16 * lh + 8];
        }
        // ---- A fragments + WMMA (16 per wave per K-step) ----
        for (int mi = 0; mi < 4; ++mi) {
            int arow = wm * 64 + mi * 16 + r;
            BF16Frag a;
            a.h[0] = *(const v8bf*)&As[arow][16 * lh];
            a.h[1] = *(const v8bf*)&As[arow][16 * lh + 8];
            for (int ni = 0; ni < 4; ++ni)
                acc[mi][ni] = wmma_bf16(a.v, bfrag[ni].v, acc[mi][ni]);
        }
        __syncthreads();
    }

    // ---- epilogue: D layout m = 8*lh + e, n = lane%16 ----
    for (int mi = 0; mi < 4; ++mi) {
        long m0 = mbase + wm * 64 + mi * 16 + 8 * lh;
        for (int ni = 0; ni < 4; ++ni) {
            int col = nbase + wn * 64 + ni * 16 + r;
            float bv = bias[col];
            for (int e = 0; e < 8; ++e) {
                float val = acc[mi][ni][e] + bv;
                long idx = (m0 + e) * (long)N + col;
                if (OUT_BF16) ((__bf16*)Yp)[idx] = (__bf16)val;
                else          ((float*)Yp)[idx]  = val;
            }
        }
    }
}

// ---------------------------------------------------------------------------
// Chunked local attention. One block per (chunk c, head hh, batch b).
// Q/K/V tiles are [128 x 64] bf16 slices of the projected [B*S, 1024] arrays.
// scores = Q K^T / 8; softmax rows; attn = P V; scatter into the torch-
// faithful concat layout: concat[b, c*128 + hh*8 + p/16, (p%16)*64 + d].
// Q/K staged via async global->LDS; V staged transposed through VGPRs.
// ---------------------------------------------------------------------------
__global__ __launch_bounds__(256)
void attn_chunk_kernel(const __bf16* __restrict__ Qb,
                       const __bf16* __restrict__ Kb,
                       const __bf16* __restrict__ Vb,
                       __bf16* __restrict__ Cc) {
    const int c  = blockIdx.x;   // chunk
    const int hh = blockIdx.y;   // head
    const int b  = blockIdx.z;   // batch

    __shared__ __attribute__((aligned(16))) __bf16 Qs[128][72];
    __shared__ __attribute__((aligned(16))) __bf16 Ks[128][72];
    __shared__ __attribute__((aligned(16))) __bf16 Vt[64][136];   // transposed V
    __shared__ __attribute__((aligned(16))) float  Sc[128][132];  // fp32 scores
    __shared__ __attribute__((aligned(16))) __bf16 Ps[128][136];  // softmax probs

    const int tid  = threadIdx.x;
    const int lane = tid & 31;
    const int wave = tid >> 5;   // 0..7
    const int r    = lane & 15;
    const int lh   = lane >> 4;

    const long rowbase = (long)b * 8192 + (long)c * P_CHUNK;
    const int  colbase = hh * D_K;

    // ---- stage Q, K (async -> LDS) and transposed V (via VGPRs) ----
    for (int pass = 0; pass < 4; ++pass) {
        int row = pass * 32 + (tid >> 3);
        int ck  = (tid & 7) * 8;
        long g  = (rowbase + row) * D_MODEL + colbase + ck;
        async_copy_b128(lds_byte_off(&Qs[row][ck]), &Qb[g]);
        async_copy_b128(lds_byte_off(&Ks[row][ck]), &Kb[g]);
        v8bf vv = *(const v8bf*)&Vb[g];
        for (int e = 0; e < 8; ++e) Vt[ck + e][row] = vv[e];
    }
    wait_asynccnt0();
    __syncthreads();

    // ---- scores: wave owns a 32(m) x 64(n) tile (2x4 accumulators) ----
    {
        const int rm = (wave & 3) * 32;
        const int cn = (wave >> 2) * 64;
        v8f sacc[2][4];
        for (int mi = 0; mi < 2; ++mi)
            for (int ni = 0; ni < 4; ++ni) sacc[mi][ni] = zero8();
        for (int ds = 0; ds < D_K; ds += 32) {
            BF16Frag bf[4];
            for (int ni = 0; ni < 4; ++ni) {
                int brow = cn + ni * 16 + r;
                bf[ni].h[0] = *(const v8bf*)&Ks[brow][ds + 16 * lh];
                bf[ni].h[1] = *(const v8bf*)&Ks[brow][ds + 16 * lh + 8];
            }
            for (int mi = 0; mi < 2; ++mi) {
                int arow = rm + mi * 16 + r;
                BF16Frag a;
                a.h[0] = *(const v8bf*)&Qs[arow][ds + 16 * lh];
                a.h[1] = *(const v8bf*)&Qs[arow][ds + 16 * lh + 8];
                for (int ni = 0; ni < 4; ++ni)
                    sacc[mi][ni] = wmma_bf16(a.v, bf[ni].v, sacc[mi][ni]);
            }
        }
        const float scale = 0.125f;  // 1/sqrt(64)
        for (int mi = 0; mi < 2; ++mi)
            for (int ni = 0; ni < 4; ++ni)
                for (int e = 0; e < 8; ++e)
                    Sc[rm + mi * 16 + 8 * lh + e][cn + ni * 16 + r] =
                        sacc[mi][ni][e] * scale;
    }
    __syncthreads();

    // ---- softmax: one thread per row, fp32 in LDS ----
    if (tid < 128) {
        float mx = -1e30f;
        for (int j = 0; j < 128; ++j) mx = fmaxf(mx, Sc[tid][j]);
        float sum = 0.f;
        for (int j = 0; j < 128; ++j) {
            float e = __expf(Sc[tid][j] - mx);
            Sc[tid][j] = e;
            sum += e;
        }
        float inv = 1.f / sum;
        for (int j = 0; j < 128; ++j)
            Ps[tid][j] = (__bf16)(Sc[tid][j] * inv);
    }
    __syncthreads();

    // ---- attn = P @ V: wave owns a 32(m) x 32(n) tile (2x2 accumulators) ----
    const int rm = (wave & 3) * 32;
    const int cn = (wave >> 2) * 32;
    v8f oacc[2][2];
    for (int mi = 0; mi < 2; ++mi)
        for (int ni = 0; ni < 2; ++ni) oacc[mi][ni] = zero8();
    for (int qs = 0; qs < 128; qs += 32) {
        BF16Frag bf[2];
        for (int ni = 0; ni < 2; ++ni) {
            int brow = cn + ni * 16 + r;
            bf[ni].h[0] = *(const v8bf*)&Vt[brow][qs + 16 * lh];
            bf[ni].h[1] = *(const v8bf*)&Vt[brow][qs + 16 * lh + 8];
        }
        for (int mi = 0; mi < 2; ++mi) {
            int arow = rm + mi * 16 + r;
            BF16Frag a;
            a.h[0] = *(const v8bf*)&Ps[arow][qs + 16 * lh];
            a.h[1] = *(const v8bf*)&Ps[arow][qs + 16 * lh + 8];
            for (int ni = 0; ni < 2; ++ni)
                oacc[mi][ni] = wmma_bf16(a.v, bf[ni].v, oacc[mi][ni]);
        }
    }

    // ---- scatter into concat layout (p = rm + mi*16 + 8*lh + e) ----
    const long cbase = (long)b * 8192 + (long)c * P_CHUNK + hh * 8;
    for (int mi = 0; mi < 2; ++mi) {
        for (int ni = 0; ni < 2; ++ni) {
            int d = cn + ni * 16 + r;
            for (int e = 0; e < 8; ++e) {
                int p = rm + mi * 16 + 8 * lh + e;
                long crow = cbase + (p >> 4);
                Cc[crow * D_MODEL + (p & 15) * D_K + d] = (__bf16)oacc[mi][ni][e];
            }
        }
    }
}

// ---------------------------------------------------------------------------
extern "C" void kernel_launch(void* const* d_in, const int* in_sizes, int n_in,
                              void* d_out, int out_size, void* d_ws, size_t ws_size,
                              hipStream_t stream) {
    (void)in_sizes; (void)n_in; (void)out_size; (void)ws_size;
    const float* q   = (const float*)d_in[0];
    const float* k   = (const float*)d_in[1];
    const float* v   = (const float*)d_in[2];
    const float* W_Q = (const float*)d_in[3];
    const float* b_Q = (const float*)d_in[4];
    const float* W_K = (const float*)d_in[5];
    const float* b_K = (const float*)d_in[6];
    const float* W_V = (const float*)d_in[7];
    const float* b_V = (const float*)d_in[8];
    const float* W_O = (const float*)d_in[9];
    const float* b_O = (const float*)d_in[10];

    const size_t MN = (size_t)NROWS * D_MODEL;
    __bf16* Qb = (__bf16*)d_ws;      // projected Q (bf16)
    __bf16* Kb = Qb + MN;            // projected K
    __bf16* Vb = Kb + MN;            // projected V
    __bf16* Cc = Vb + MN;            // attention output in concat layout

    dim3 gGemm(D_MODEL / 256, NROWS / 128);  // (4, 256)
    dim3 blk(256);

    gemm_bias_kernel<false, true><<<gGemm, blk, 0, stream>>>(q, W_Q, b_Q, Qb);
    gemm_bias_kernel<false, true><<<gGemm, blk, 0, stream>>>(k, W_K, b_K, Kb);
    gemm_bias_kernel<false, true><<<gGemm, blk, 0, stream>>>(v, W_V, b_V, Vb);

    attn_chunk_kernel<<<dim3(64, 16, 4), blk, 0, stream>>>(Qb, Kb, Vb, Cc);

    gemm_bias_kernel<true, false><<<gGemm, blk, 0, stream>>>(Cc, W_O, b_O, d_out);
}
